// TotalLossFunc_2138893713776
// MI455X (gfx1250) — matrix operand order repaired and checked by reading, not compiled, not run on hardware
//
#include <hip/hip_runtime.h>
#include <hip/hip_bf16.h>

// ---------------------------------------------------------------------------
// Problem constants (from the reference)
// ---------------------------------------------------------------------------
#define N_ROWS 16384
#define DDIM   2048
#define C_RAW  1000
#define CP     1024   // padded class count for the GEMM

// GEMM tiling
#define TM   128
#define TN   128
#define TK   32
#define LSTR 40       // LDS row stride in halfs (32 + 8 pad) = 80B = 5 * 16B
#define KSTEPS (DDIM / TK)

// TDM builtin arity: the 6-arg toolchain is exactly the one shipping the
// gfx1250 TDM header; ROCm 7.2 (clang-22) has the 5-arg form and no header.
// __has_include is preprocessor-stable across host and device passes.
#if __has_include(<hip/amd_detail/amd_gfx1250_TDM.h>)
#define TDM_ARITY 6
#else
#define TDM_ARITY 5
#endif

typedef __attribute__((ext_vector_type(16))) __bf16 v16bf;
typedef __attribute__((ext_vector_type(8)))  float  v8f;
typedef __attribute__((ext_vector_type(4)))  unsigned v4u;
typedef __attribute__((ext_vector_type(8)))  int      v8i;
typedef __attribute__((ext_vector_type(4)))  int      v4i;

union Frag { v16bf v; uint4 q[2]; };
union H8   { __bf16 h[8]; uint4 q; };

// ---------------------------------------------------------------------------
// Phase 1: f32 -> bf16 (hi + residual lo) conversion + row sum-of-squares
// ---------------------------------------------------------------------------
template<bool SPLIT>
__global__ __launch_bounds__(256) void prep_x_kernel(
    const float* __restrict__ X, __bf16* __restrict__ Xhi,
    __bf16* __restrict__ Xlo, float* __restrict__ xsq)
{
    const int n   = blockIdx.x;
    const int tid = threadIdx.x;
    const size_t base = (size_t)n * DDIM + (size_t)tid * 8;

    H8 hi, lo;
    float s = 0.f;
#pragma unroll
    for (int i = 0; i < 8; ++i) {
        float v = X[base + i];
        s += v * v;
        __bf16 h = (__bf16)v;
        hi.h[i] = h;
        if (SPLIT) lo.h[i] = (__bf16)(v - (float)h);
    }
    *(uint4*)(Xhi + base) = hi.q;
    if (SPLIT) *(uint4*)(Xlo + base) = lo.q;

    __shared__ float red[256];
    red[tid] = s;
    __syncthreads();
    for (int o = 128; o > 0; o >>= 1) {
        if (tid < o) red[tid] += red[tid + o];
        __syncthreads();
    }
    if (tid == 0) xsq[n] = red[0];
}

template<bool SPLIT>
__global__ __launch_bounds__(256) void prep_mu_kernel(
    const float* __restrict__ MU, __bf16* __restrict__ Mhi,
    __bf16* __restrict__ Mlo, float* __restrict__ musq)
{
    const int c   = blockIdx.x;   // 0..CP-1 (padded)
    const int tid = threadIdx.x;
    const size_t obase = (size_t)c * DDIM + (size_t)tid * 8;

    H8 hi, lo;
    float s = 0.f;
    if (c < C_RAW) {
#pragma unroll
        for (int i = 0; i < 8; ++i) {
            float v = MU[obase + i];
            s += v * v;
            __bf16 h = (__bf16)v;
            hi.h[i] = h;
            if (SPLIT) lo.h[i] = (__bf16)(v - (float)h);
        }
    } else {
#pragma unroll
        for (int i = 0; i < 8; ++i) {
            hi.h[i] = (__bf16)0.f;
            if (SPLIT) lo.h[i] = (__bf16)0.f;
        }
    }
    *(uint4*)(Mhi + obase) = hi.q;
    if (SPLIT) *(uint4*)(Mlo + obase) = lo.q;

    __shared__ float red[256];
    red[tid] = s;
    __syncthreads();
    for (int o = 128; o > 0; o >>= 1) {
        if (tid < o) red[tid] += red[tid + o];
        __syncthreads();
    }
    if (tid == 0) musq[c] = red[0];
}

// ---------------------------------------------------------------------------
// TDM descriptor issue: load a TM x TK bf16 tile (row stride DDIM elements)
// into LDS with hardware padding (16 DWORDs data -> skip 4 DWORDs) so each
// 32-half tile row lands at an 80-byte (LSTR) LDS stride.
// D# packing per CDNA5 ISA 8.3/8.4 (2-D tensor; groups 2/3 zero).
// ---------------------------------------------------------------------------
__device__ __forceinline__ unsigned lds_byte_off(const void* p)
{
    // Flat-aperture rule: LDS offset is the low 32 bits of the generic address
    return (unsigned)(unsigned long long)(uintptr_t)p;
}

__device__ __forceinline__ void tdm_load_tile(const __bf16* gtile,
                                              unsigned ldsAddr,
                                              unsigned tensorRows)
{
    const unsigned long long ga = (unsigned long long)(uintptr_t)gtile;
    v4u g0;
    g0[0] = 1u;                                        // count=1, user D#
    g0[1] = ldsAddr;                                   // lds_addr [63:32]
    g0[2] = (unsigned)(ga & 0xffffffffu);              // global_addr lo
    g0[3] = (unsigned)((ga >> 32) & 0x01ffffffu)       // global_addr[56:32]
          | (2u << 30);                                // type=2 ("image")
    v8i g1;
    g1[0] = (int)((1u << 16)       // data_size = 2 bytes
                | (1u << 20)       // pad_enable
                | (3u << 22)       // pad_interval: 16 DWORDs (64B = tile row)
                | (3u << 25));     // pad_amount:   4 DWORDs (16B pad -> 80B)
    g1[1] = (int)(((unsigned)DDIM & 0xffffu) << 16);   // tensor_dim0 lo16
    g1[2] = (int)((((unsigned)DDIM >> 16) & 0xffffu)   // tensor_dim0 hi16
                | ((tensorRows & 0xffffu) << 16));     // tensor_dim1 lo16
    g1[3] = (int)(((tensorRows >> 16) & 0xffffu)       // tensor_dim1 hi16
                | ((unsigned)TK << 16));               // tile_dim0 = 32
    g1[4] = (int)TM;                                   // tile_dim1=128, tile_dim2=0
    g1[5] = (int)DDIM;                                 // tensor_dim0_stride lo32
    g1[6] = 0;                                         // stride hi16 | dim1_stride lo
    g1[7] = 0;                                         // dim1_stride hi
    const v4i z4 = {0, 0, 0, 0};
#if TDM_ARITY == 6
    const v8i z8 = {0, 0, 0, 0, 0, 0, 0, 0};
    __builtin_amdgcn_tensor_load_to_lds(g0, g1, z4, z4, z8, 0);
#else
    __builtin_amdgcn_tensor_load_to_lds(g0, g1, z4, z4, 0);
#endif
}

// ---------------------------------------------------------------------------
// Phase 2: cross[N, CP] = X * Mu^T via v_wmma_f32_16x16x32_bf16
//   block = 128x128 tile, 256 threads = 8 waves; wave tile = 32x64 (2x4 WMMA)
//   split path: cross ~= hi*hi + hi*lo + lo*hi  (3 WMMAs per 16x16x32 step)
//   Staging: TDM double-buffered (wave 0 issues, TENSORcnt + barrier publish)
// ---------------------------------------------------------------------------
template<bool SPLIT>
__global__ __launch_bounds__(256) void gemm_wmma_tdm_kernel(
    const __bf16* __restrict__ Xhi, const __bf16* __restrict__ Xlo,
    const __bf16* __restrict__ Mhi, const __bf16* __restrict__ Mlo,
    float* __restrict__ crossOut)
{
    __shared__ __bf16 sXh[2][TM * LSTR];
    __shared__ __bf16 sMh[2][TN * LSTR];
    __shared__ __bf16 sXl[SPLIT ? 2 : 1][TM * LSTR];
    __shared__ __bf16 sMl[SPLIT ? 2 : 1][TN * LSTR];

    const int tid  = threadIdx.x;
    const int lane = tid & 31;
    const int w    = tid >> 5;
    const int wr   = (w & 3) * 32;   // wave row offset inside block tile
    const int wc   = (w >> 2) * 64;  // wave col offset inside block tile
    const int bn   = blockIdx.y * TM;
    const int bc   = blockIdx.x * TN;

    const v8f vzero = {0.f, 0.f, 0.f, 0.f, 0.f, 0.f, 0.f, 0.f};
    v8f acc[2][4];
#pragma unroll
    for (int m = 0; m < 2; ++m)
#pragma unroll
        for (int n = 0; n < 4; ++n) acc[m][n] = vzero;

    // ISA 7.12.2 lane layouts (wave32):
    //  A 16x32 bf16: lanes 0-15 -> K chunks {0..7, 16..23}; lanes 16-31 -> {8..15, 24..31}
    //  B 32x16 bf16: lanes 0-15 -> K chunks {0..7, 16..23}; lanes 16-31 -> {8..15, 24..31}
    const int c0  = (lane < 16) ? 0 : 8;
    const int kb  = (lane < 16) ? 0 : 16;
    const int l15 = lane & 15;

    const bool issuer = (tid < 32);   // wave 0 drives the TDM

    if (issuer) {
        tdm_load_tile(Xhi + (size_t)bn * DDIM, lds_byte_off(&sXh[0][0]), N_ROWS);
        tdm_load_tile(Mhi + (size_t)bc * DDIM, lds_byte_off(&sMh[0][0]), CP);
        if (SPLIT) {
            tdm_load_tile(Xlo + (size_t)bn * DDIM, lds_byte_off(&sXl[0][0]), N_ROWS);
            tdm_load_tile(Mlo + (size_t)bc * DDIM, lds_byte_off(&sMl[0][0]), CP);
        }
    }

    for (int it = 0; it < KSTEPS; ++it) {
        if (issuer) __builtin_amdgcn_s_wait_tensorcnt(0);
        __syncthreads();   // tile `it` visible to all waves; buffer it^1 free

        if (issuer && (it + 1) < KSTEPS) {
            const int k1 = (it + 1) * TK;
            const int nb = (it + 1) & 1;
            tdm_load_tile(Xhi + (size_t)bn * DDIM + k1, lds_byte_off(&sXh[nb][0]), N_ROWS);
            tdm_load_tile(Mhi + (size_t)bc * DDIM + k1, lds_byte_off(&sMh[nb][0]), CP);
            if (SPLIT) {
                tdm_load_tile(Xlo + (size_t)bn * DDIM + k1, lds_byte_off(&sXl[nb][0]), N_ROWS);
                tdm_load_tile(Mlo + (size_t)bc * DDIM + k1, lds_byte_off(&sMl[nb][0]), CP);
            }
        }

        const __bf16* bXh = sXh[it & 1];
        const __bf16* bMh = sMh[it & 1];
        const __bf16* bXl = sXl[SPLIT ? (it & 1) : 0];
        const __bf16* bMl = sMl[SPLIT ? (it & 1) : 0];

        Frag ah[2], al[2], bh[4], bl[4];
#pragma unroll
        for (int m = 0; m < 2; ++m) {
            const int ro = (wr + m * 16 + l15) * LSTR;
            ah[m].q[0] = *(const uint4*)(bXh + ro + c0);
            ah[m].q[1] = *(const uint4*)(bXh + ro + c0 + 16);
            if (SPLIT) {
                al[m].q[0] = *(const uint4*)(bXl + ro + c0);
                al[m].q[1] = *(const uint4*)(bXl + ro + c0 + 16);
            }
        }
#pragma unroll
        for (int n = 0; n < 4; ++n) {
            const int ro = (wc + n * 16 + l15) * LSTR + c0;
            bh[n].q[0] = *(const uint4*)(bMh + ro);
            bh[n].q[1] = *(const uint4*)(bMh + ro + 16);
            if (SPLIT) {
                bl[n].q[0] = *(const uint4*)(bMl + ro);
                bl[n].q[1] = *(const uint4*)(bMl + ro + 16);
            }
        }
#pragma unroll
        for (int m = 0; m < 2; ++m) {
#pragma unroll
            for (int n = 0; n < 4; ++n) {
                acc[m][n] = __builtin_amdgcn_wmma_f32_16x16x32_bf16(
                    false, ah[m].v, false, bh[n].v, (short)0, acc[m][n], false, false);
                if (SPLIT) {
                    acc[m][n] = __builtin_amdgcn_wmma_f32_16x16x32_bf16(
                        false, ah[m].v, false, bl[n].v, (short)0, acc[m][n], false, false);
                    acc[m][n] = __builtin_amdgcn_wmma_f32_16x16x32_bf16(
                        false, al[m].v, false, bh[n].v, (short)0, acc[m][n], false, false);
                }
            }
        }
    }

    // Epilogue: C/D layout -> VGPR j holds M=j (lanes 0-15) / M=j+8 (lanes 16-31)
    const int rbase = bn + wr + ((lane >> 4) << 3);
    const int cbase = bc + wc + l15;
#pragma unroll
    for (int m = 0; m < 2; ++m)
#pragma unroll
        for (int n = 0; n < 4; ++n)
#pragma unroll
            for (int j = 0; j < 8; ++j)
                crossOut[(size_t)(rbase + m * 16 + j) * CP + (cbase + n * 16)] =
                    acc[m][n][j];
}

// ---------------------------------------------------------------------------
// Phase 3: per-row distance -> adjust -> log-softmax NLL (one block per row)
// ---------------------------------------------------------------------------
__global__ __launch_bounds__(128) void loss_rows_kernel(
    const float* __restrict__ cross, const float* __restrict__ xsq,
    const float* __restrict__ musq, const int* __restrict__ y,
    float* __restrict__ lossN)
{
    const int n   = blockIdx.x;
    const int tid = threadIdx.x;
    __shared__ float sA[CP];
    __shared__ float red[128];

    const float xs  = xsq[n];
    const int   lbl = y[n];
    const float inf = __builtin_inff();

    for (int c = tid; c < CP; c += 128) {
        float v;
        if (c < C_RAW) {
            float sq = fmaxf(xs - 2.0f * cross[(size_t)n * CP + c] + musq[c], 0.f);
            float d  = sqrtf(0.5f * sq);
            v = (c == lbl) ? d * 1.5f : d;   // * (1 + ALPHA*onehot), ALPHA=0.5
        } else {
            v = inf;                          // padded class: z = -inf
        }
        sA[c] = v;
    }
    __syncthreads();

    // max of z = -a  <=>  min of a
    float mn = inf;
    for (int c = tid; c < CP; c += 128) mn = fminf(mn, sA[c]);
    red[tid] = mn;
    __syncthreads();
    for (int o = 64; o > 0; o >>= 1) {
        if (tid < o) red[tid] = fminf(red[tid], red[tid + o]);
        __syncthreads();
    }
    const float m = red[0];
    __syncthreads();

    float s = 0.f;
    for (int c = tid; c < C_RAW; c += 128) s += expf(m - sA[c]);
    red[tid] = s;
    __syncthreads();
    for (int o = 64; o > 0; o >>= 1) {
        if (tid < o) red[tid] += red[tid + o];
        __syncthreads();
    }
    if (tid == 0) lossN[n] = sA[lbl] - m + logf(red[0]);   // a_y + logsumexp(-a)
}

// ---------------------------------------------------------------------------
// Phase 4: deterministic mean over rows -> scalar output
// ---------------------------------------------------------------------------
__global__ __launch_bounds__(256) void final_reduce_kernel(
    const float* __restrict__ lossN, float* __restrict__ out)
{
    __shared__ float red[256];
    const int tid = threadIdx.x;
    float s = 0.f;
    for (int i = tid; i < N_ROWS; i += 256) s += lossN[i];
    red[tid] = s;
    __syncthreads();
    for (int o = 128; o > 0; o >>= 1) {
        if (tid < o) red[tid] += red[tid + o];
        __syncthreads();
    }
    if (tid == 0) out[0] = red[0] / (float)N_ROWS;
}

// ---------------------------------------------------------------------------
// Host-side launch
// ---------------------------------------------------------------------------
static inline size_t alignup(size_t x) { return (x + 255) & ~(size_t)255; }

extern "C" void kernel_launch(void* const* d_in, const int* in_sizes, int n_in,
                              void* d_out, int out_size, void* d_ws, size_t ws_size,
                              hipStream_t stream)
{
    (void)in_sizes; (void)n_in; (void)out_size;
    const float* X  = (const float*)d_in[0];
    const int*   Y  = (const int*)d_in[1];
    const float* MU = (const float*)d_in[2];

    const size_t SZ_XH  = (size_t)N_ROWS * DDIM * sizeof(__bf16);
    const size_t SZ_MH  = (size_t)CP     * DDIM * sizeof(__bf16);
    const size_t SZ_CR  = (size_t)N_ROWS * CP   * sizeof(float);
    const size_t SZ_XSQ = (size_t)N_ROWS * sizeof(float);
    const size_t SZ_MSQ = (size_t)CP     * sizeof(float);
    const size_t SZ_LN  = (size_t)N_ROWS * sizeof(float);

    const size_t need_base  = alignup(SZ_XH) + alignup(SZ_MH) + alignup(SZ_CR) +
                              alignup(SZ_XSQ) + alignup(SZ_MSQ) + alignup(SZ_LN);
    const size_t need_split = need_base + alignup(SZ_XH) + alignup(SZ_MH);
    const bool split = (ws_size >= need_split);   // deterministic (size-only)

    size_t off = 0;
    char* base = (char*)d_ws;
    auto take = [&](size_t bytes) { char* p = base + off; off += alignup(bytes); return p; };

    __bf16* Xhi   = (__bf16*)take(SZ_XH);
    __bf16* Mhi   = (__bf16*)take(SZ_MH);
    float*  crossW= (float*) take(SZ_CR);
    float*  xsq   = (float*) take(SZ_XSQ);
    float*  musq  = (float*) take(SZ_MSQ);
    float*  lossN = (float*) take(SZ_LN);
    __bf16* Xlo   = nullptr;
    __bf16* Mlo   = nullptr;
    if (split) {
        Xlo = (__bf16*)take(SZ_XH);
        Mlo = (__bf16*)take(SZ_MH);
    }

    dim3 gGemm(CP / TN, N_ROWS / TM);
    if (split) {
        prep_x_kernel<true><<<N_ROWS, 256, 0, stream>>>(X, Xhi, Xlo, xsq);
        prep_mu_kernel<true><<<CP, 256, 0, stream>>>(MU, Mhi, Mlo, musq);
        gemm_wmma_tdm_kernel<true><<<gGemm, 256, 0, stream>>>(Xhi, Xlo, Mhi, Mlo, crossW);
    } else {
        prep_x_kernel<false><<<N_ROWS, 256, 0, stream>>>(X, Xhi, nullptr, xsq);
        prep_mu_kernel<false><<<CP, 256, 0, stream>>>(MU, Mhi, nullptr, musq);
        gemm_wmma_tdm_kernel<false><<<gGemm, 256, 0, stream>>>(Xhi, nullptr, Mhi, nullptr, crossW);
    }
    loss_rows_kernel<<<N_ROWS, 128, 0, stream>>>(crossW, xsq, musq, Y, lossN);
    final_reduce_kernel<<<1, 256, 0, stream>>>(lossN, (float*)d_out);
}
